// Attention_74302934220843
// MI455X (gfx1250) — matrix operand order repaired and checked by reading, not compiled, not run on hardware
//
#include <hip/hip_runtime.h>
#include <hip/hip_bf16.h>

// ---------------------------------------------------------------------------
// Types
// ---------------------------------------------------------------------------
typedef __bf16 bf16;
typedef __attribute__((ext_vector_type(16))) __bf16 bf16x16;
typedef __attribute__((ext_vector_type(8)))  float  f32x8;
typedef __attribute__((ext_vector_type(4)))  unsigned int u32x4;
typedef __attribute__((ext_vector_type(8)))  int    i32x8;
typedef __attribute__((ext_vector_type(4)))  int    i32x4;

#define DIM_   2048
#define NH_    16
#define NKV_   4
#define HD_    128
#define BS_    2
#define SEQ_   4096
#define MROWS_ (BS_ * SEQ_)   // 8192

// ---------------------------------------------------------------------------
// Tensor Data Mover: 2-D tile (global -> LDS) with LDS row padding.
// D# packing per CDNA5 ISA ch. "Tensor DMA Descriptor".
//   data_size      = 2 bytes (bf16)
//   tile_w elems/row, tile_h rows, global row stride in elements
//   pad_interval_code: pad after 2^(c+1) DWORDs ; pad_amount_code: c+1 DWORDs
// lds byte address = low 32 bits of the generic pointer (ISA: addr[31:0]).
// ---------------------------------------------------------------------------
__device__ inline void tdm_load_2d(const void* gptr, const void* lptr,
                                   unsigned tile_w, unsigned tile_h,
                                   unsigned long long row_stride_elems,
                                   unsigned pad_interval_code,
                                   unsigned pad_amount_code) {
    const unsigned long long ga = (unsigned long long)gptr;
    const unsigned lds = (unsigned)(unsigned long long)lptr;
    u32x4 g0;
    g0.x = 1u;                                   // count=1 (valid), user mode
    g0.y = lds;                                  // lds_addr (bytes)
    g0.z = (unsigned)ga;                         // global_addr[31:0]
    g0.w = (unsigned)((ga >> 32) & 0x1ffffffull) // global_addr[56:32]
           | (2u << 30);                         // type = 2 ("image")
    const unsigned td0 = 0x40000000u;            // huge tensor dims: no OOB clip
    const unsigned td1 = 0x40000000u;
    i32x8 g1;
    g1[0] = (int)((1u << 16)                     // data_size = 2 bytes
                  | (1u << 20)                   // pad_enable
                  | (pad_interval_code << 22)
                  | (pad_amount_code << 25));
    g1[1] = (int)((td0 & 0xffffu) << 16);        // abar=0 | tensor_dim0[15:0]
    g1[2] = (int)((td0 >> 16) | ((td1 & 0xffffu) << 16));
    g1[3] = (int)((td1 >> 16) | (tile_w << 16)); // tensor_dim1[31:16] | tile_dim0
    g1[4] = (int)tile_h;                         // tile_dim1 | tile_dim2=0
    g1[5] = (int)(unsigned)row_stride_elems;     // tensor_dim0_stride[31:0]
    g1[6] = (int)(unsigned)((row_stride_elems >> 32) & 0xffffull);
    g1[7] = 0;                                   // tensor_dim1_stride unused (2D)
    i32x4 z4 = {0, 0, 0, 0};
#if defined(__clang_major__) && __clang_major__ >= 23
    i32x8 z8 = {0, 0, 0, 0, 0, 0, 0, 0};
    __builtin_amdgcn_tensor_load_to_lds(g0, g1, z4, z4, z8, 0);
#else
    __builtin_amdgcn_tensor_load_to_lds(g0, g1, z4, z4, 0);
#endif
}

// ---------------------------------------------------------------------------
// fp32 -> bf16 conversion (grid-stride)
// ---------------------------------------------------------------------------
__global__ void cvt_f32_to_bf16(const float* __restrict__ s,
                                bf16* __restrict__ d, long long n) {
    long long i = (long long)blockIdx.x * blockDim.x + threadIdx.x;
    long long stride = (long long)gridDim.x * blockDim.x;
    for (; i < n; i += stride) d[i] = (bf16)s[i];
}

// ---------------------------------------------------------------------------
// Fragment loaders per CDNA5 WMMA VGPR layouts (16x16x32 bf16).
// A (16x32, MxK): lane L covers row M = L&15; first 8 elems = K[(L>>4)*8 ..],
//                 second 8 elems = K[16 + (L>>4)*8 ..].
// B (32x16, KxN): lane L covers col N = L&15; same K order as A.
//                 Source stored row-major as [N][K].
// C/D (16x16):    lane L, elem e -> row M = e + 8*(L>>4), col N = L&15.
// ---------------------------------------------------------------------------
__device__ inline bf16x16 load_a_frag(const bf16* __restrict__ p, int ld, int lane) {
    const int half = lane >> 4;
    const int r    = lane & 15;
    const bf16* row = p + r * ld;
    bf16x16 f;
#pragma unroll
    for (int i = 0; i < 8; ++i) f[i]     = row[half * 8 + i];
#pragma unroll
    for (int i = 0; i < 8; ++i) f[i + 8] = row[16 + half * 8 + i];
    return f;
}

__device__ inline bf16x16 load_bT_frag(const bf16* __restrict__ p, int ld, int lane) {
    const int half = lane >> 4;
    const int n    = lane & 15;
    const bf16* row = p + n * ld;
    bf16x16 f;
#pragma unroll
    for (int i = 0; i < 8; ++i) f[i]     = row[half * 8 + i];
#pragma unroll
    for (int i = 0; i < 8; ++i) f[i + 8] = row[16 + half * 8 + i];
    return f;
}

// ---------------------------------------------------------------------------
// Tiled WMMA GEMM:  C[M,N] = A[M,K] * W[N,K]^T
// Block tile 128x128, 256 threads = 8 waves (2(M) x 4(N)), each wave 64x32.
// Double-buffered LDS tiles filled by the Tensor Data Mover: wave 0 issues
// the TDM pair for tile k+1, waits tensorcnt<=2 (pair k done, in-order),
// then the block computes on tile k while the TDM streams tile k+1.
// ---------------------------------------------------------------------------
template <bool OUT_BF16>
__global__ __launch_bounds__(256) void gemm_nt_bf16(
    const bf16* __restrict__ A, const bf16* __restrict__ W,
    void* __restrict__ Cout, int M, int N, int K) {
    constexpr int BM = 128, BN = 128, BK = 32;
    constexpr int LDT = BK + 16;  // 48 elems: row 64B + 32B pad (TDM pad codes 3/7)
    __shared__ bf16 As[2][BM * LDT];
    __shared__ bf16 Ws[2][BN * LDT];

    const int tid  = threadIdx.x;
    const int wid  = tid >> 5;
    const int lane = tid & 31;
    const int wm   = wid >> 2;     // 0..1
    const int wn   = wid & 3;      // 0..3
    const int m0   = blockIdx.y * BM;
    const int n0   = blockIdx.x * BN;

    f32x8 acc[4][2] = {};
    const int nk = K / BK;

    if (wid == 0) {
        tdm_load_2d(A + (size_t)m0 * K, As[0], BK, BM, (unsigned long long)K, 3u, 7u);
        tdm_load_2d(W + (size_t)n0 * K, Ws[0], BK, BN, (unsigned long long)K, 3u, 7u);
    }

    for (int kb = 0; kb < nk; ++kb) {
        const int cur = kb & 1;
        if (wid == 0) {
            if (kb + 1 < nk) {
                const int k1 = (kb + 1) * BK;
                tdm_load_2d(A + (size_t)m0 * K + k1, As[1 - cur], BK, BM,
                            (unsigned long long)K, 3u, 7u);
                tdm_load_2d(W + (size_t)n0 * K + k1, Ws[1 - cur], BK, BN,
                            (unsigned long long)K, 3u, 7u);
                __builtin_amdgcn_s_wait_tensorcnt(2);
            } else {
                __builtin_amdgcn_s_wait_tensorcnt(0);
            }
        }
        __syncthreads();

        const bf16* as = As[cur];
        const bf16* ws = Ws[cur];
        bf16x16 bfrag[2];
#pragma unroll
        for (int j = 0; j < 2; ++j)
            bfrag[j] = load_bT_frag(ws + (wn * 32 + j * 16) * LDT, LDT, lane);
#pragma unroll
        for (int i = 0; i < 4; ++i) {
            bf16x16 afrag = load_a_frag(as + (wm * 64 + i * 16) * LDT, LDT, lane);
#pragma unroll
            for (int j = 0; j < 2; ++j)
                acc[i][j] = __builtin_amdgcn_wmma_f32_16x16x32_bf16(
                    false, afrag, false, bfrag[j], (short)0, acc[i][j], false, false);
        }
        __syncthreads();
    }

    const int half = lane >> 4, nl = lane & 15;
#pragma unroll
    for (int i = 0; i < 4; ++i)
#pragma unroll
        for (int j = 0; j < 2; ++j) {
            const int col = n0 + wn * 32 + j * 16 + nl;
#pragma unroll
            for (int e = 0; e < 8; ++e) {
                const int row = m0 + wm * 64 + i * 16 + e + 8 * half;
                if (OUT_BF16)
                    ((bf16*)Cout)[(size_t)row * N + col] = (bf16)acc[i][j][e];
                else
                    ((float*)Cout)[(size_t)row * N + col] = acc[i][j][e];
            }
        }
}

// ---------------------------------------------------------------------------
// Fused per-head RMSNorm + RoPE.  One 128-elem head per 64-thread block.
// ---------------------------------------------------------------------------
__global__ __launch_bounds__(64) void rmsnorm_rope_kernel(
    const bf16* __restrict__ in, const float* __restrict__ w,
    const float* __restrict__ freqs, bf16* __restrict__ out, int H) {
    const int blk = blockIdx.x;
    const int m = blk / H, h = blk % H;
    const int s = m % SEQ_;
    const size_t base = (size_t)m * (H * HD_) + h * HD_;
    const int i = threadIdx.x;  // 0..63 -> pair (2i, 2i+1)

    float xr = (float)in[base + 2 * i];
    float xi = (float)in[base + 2 * i + 1];
    float ss = xr * xr + xi * xi;
#pragma unroll
    for (int off = 16; off >= 1; off >>= 1) ss += __shfl_xor(ss, off, 32);
    __shared__ float red[2];
    if ((i & 31) == 0) red[i >> 5] = ss;
    __syncthreads();
    const float inv = rsqrtf((red[0] + red[1]) * (1.0f / 128.0f) + 1e-6f);

    xr = xr * inv * w[2 * i];
    xi = xi * inv * w[2 * i + 1];
    const float c  = freqs[(s * 64 + i) * 2 + 0];
    const float sn = freqs[(s * 64 + i) * 2 + 1];
    out[base + 2 * i]     = (bf16)(xr * c - xi * sn);
    out[base + 2 * i + 1] = (bf16)(xr * sn + xi * c);
}

// ---------------------------------------------------------------------------
// Flash attention (causal, GQA 16q/4kv heads, head_dim 128).
// Grid: (SEQ/128, NH, BS). Block: 256 threads = 8 waves; wave w owns q rows
// [q0+16w, q0+16w+16). K tile (32x128, pad codes 5/3) streamed by the TDM;
// V tile transposed by VALU (TDM has no transpose). Online softmax in f32.
// ---------------------------------------------------------------------------
__global__ __launch_bounds__(256) void flash_attn_kernel(
    const bf16* __restrict__ Q,   // [M, NH*HD]
    const bf16* __restrict__ Kc,  // [M, NKV*HD]
    const bf16* __restrict__ Vc,  // [M, NKV*HD]
    bf16* __restrict__ O) {       // [M, NH*HD]
    constexpr int BQ = 128, BKV = 32;
    constexpr int LDK = HD_ + 8;   // 136 elems: row 256B + 16B pad (TDM codes 5/3)
    constexpr int LDV = BKV + 8;
    constexpr int LDP = BKV + 8;
    __shared__ bf16 Ks[BKV * LDK];
    __shared__ bf16 Vt[HD_ * LDV];
    __shared__ bf16 Pb[8 * 16 * LDP];

    const int tid = threadIdx.x, wid = tid >> 5, lane = tid & 31;
    const int half = lane >> 4, nl = lane & 15;
    const int b = blockIdx.z, h = blockIdx.y, kvh = h >> 2;
    const int q0 = blockIdx.x * BQ;
    const size_t rowQ = (size_t)(b * SEQ_ + q0 + wid * 16);
    const float scale = 0.08838834764831845f;  // 1/sqrt(128)

    // Preload Q fragments (A layout) straight from global.
    bf16x16 qf[4];
#pragma unroll
    for (int dc = 0; dc < 4; ++dc) {
        const bf16* qrow = Q + (rowQ + nl) * (NH_ * HD_) + h * HD_ + dc * 32;
        bf16x16 f;
#pragma unroll
        for (int i = 0; i < 8; ++i) f[i]     = qrow[half * 8 + i];
#pragma unroll
        for (int i = 0; i < 8; ++i) f[i + 8] = qrow[16 + half * 8 + i];
        qf[dc] = f;
    }

    f32x8 oacc[8] = {};
    float mrow[8], lrow[8];
#pragma unroll
    for (int e = 0; e < 8; ++e) { mrow[e] = -3.0e38f; lrow[e] = 0.0f; }

    const int kv_end = q0 + BQ;  // causal bound (exclusive)
    for (int kv0 = 0; kv0 < kv_end; kv0 += BKV) {
        // K tile via Tensor Data Mover; V tile transposed by hand.
        if (wid == 0)
            tdm_load_2d(Kc + (size_t)(b * SEQ_ + kv0) * (NKV_ * HD_) + kvh * HD_,
                        Ks, HD_, BKV, (unsigned long long)(NKV_ * HD_), 5u, 3u);
        for (int idx = tid; idx < BKV * HD_; idx += 256) {
            const int r = idx >> 7, d = idx & 127;
            const size_t grow =
                (size_t)(b * SEQ_ + kv0 + r) * (NKV_ * HD_) + kvh * HD_ + d;
            Vt[d * LDV + r] = Vc[grow];
            if (kv0 + BKV < kv_end)
                __builtin_prefetch(Vc + grow + (size_t)BKV * (NKV_ * HD_), 0, 1);
        }
        if (wid == 0) __builtin_amdgcn_s_wait_tensorcnt(0);
        __syncthreads();

        // S = Q K^T  (16 q rows x 32 kv cols per wave)
        f32x8 sacc[2] = {};
#pragma unroll
        for (int dc = 0; dc < 4; ++dc)
#pragma unroll
            for (int j = 0; j < 2; ++j) {
                bf16x16 bf = load_bT_frag(Ks + (j * 16) * LDK + dc * 32, LDK, lane);
                sacc[j] = __builtin_amdgcn_wmma_f32_16x16x32_bf16(
                    false, qf[dc], false, bf, (short)0, sacc[j], false, false);
            }

        // Scale + causal mask + online softmax update.
        float alpha[8];
#pragma unroll
        for (int e = 0; e < 8; ++e) {
            const int qrow = q0 + wid * 16 + e + 8 * half;
            float v0 = sacc[0][e] * scale;
            float v1 = sacc[1][e] * scale;
            if (kv0 + nl > qrow)      v0 = -3.0e38f;
            if (kv0 + 16 + nl > qrow) v1 = -3.0e38f;
            float mx = fmaxf(v0, v1);
#pragma unroll
            for (int off = 8; off >= 1; off >>= 1)
                mx = fmaxf(mx, __shfl_xor(mx, off, 32));
            const float mnew = fmaxf(mrow[e], mx);
            alpha[e] = __expf(mrow[e] - mnew);
            const float p0 = __expf(v0 - mnew);
            const float p1 = __expf(v1 - mnew);
            sacc[0][e] = p0;
            sacc[1][e] = p1;
            float ps = p0 + p1;
#pragma unroll
            for (int off = 8; off >= 1; off >>= 1)
                ps += __shfl_xor(ps, off, 32);
            mrow[e] = mnew;
            lrow[e] = lrow[e] * alpha[e] + ps;
        }
#pragma unroll
        for (int j2 = 0; j2 < 8; ++j2)
#pragma unroll
            for (int e = 0; e < 8; ++e) oacc[j2][e] *= alpha[e];

        // Repack P (C layout -> row-major 16x32 bf16) via per-wave LDS buffer.
        bf16* pw = Pb + wid * 16 * LDP;
#pragma unroll
        for (int j = 0; j < 2; ++j)
#pragma unroll
            for (int e = 0; e < 8; ++e)
                pw[(e + 8 * half) * LDP + j * 16 + nl] = (bf16)sacc[j][e];
        asm volatile("s_wait_dscnt 0x0" ::: "memory");

        // O += P * V
        bf16x16 pf = load_a_frag(pw, LDP, lane);
#pragma unroll
        for (int j2 = 0; j2 < 8; ++j2) {
            bf16x16 vf = load_bT_frag(Vt + (j2 * 16) * LDV, LDV, lane);
            oacc[j2] = __builtin_amdgcn_wmma_f32_16x16x32_bf16(
                false, pf, false, vf, (short)0, oacc[j2], false, false);
        }
        __syncthreads();
    }

    // Finalize: divide by l, store bf16.
#pragma unroll
    for (int j2 = 0; j2 < 8; ++j2)
#pragma unroll
        for (int e = 0; e < 8; ++e) {
            const size_t row = rowQ + e + 8 * half;
            O[row * (NH_ * HD_) + h * HD_ + j2 * 16 + nl] =
                (bf16)(oacc[j2][e] / lrow[e]);
        }
}

// ---------------------------------------------------------------------------
// Host side
// ---------------------------------------------------------------------------
extern "C" void kernel_launch(void* const* d_in, const int* in_sizes, int n_in,
                              void* d_out, int out_size, void* d_ws, size_t ws_size,
                              hipStream_t stream) {
    (void)in_sizes; (void)n_in; (void)out_size; (void)ws_size;
    const float* x   = (const float*)d_in[0];
    const float* fr  = (const float*)d_in[1];
    const float* wq  = (const float*)d_in[2];
    const float* wk  = (const float*)d_in[3];
    const float* wv  = (const float*)d_in[4];
    const float* wo  = (const float*)d_in[5];
    const float* qnw = (const float*)d_in[6];
    const float* knw = (const float*)d_in[7];
    float* out = (float*)d_out;

    const int M = MROWS_;
    char* wsp = (char*)d_ws;
    auto alloc = [&](size_t bytes) -> char* {
        char* p = wsp;
        wsp += (bytes + 255) & ~(size_t)255;
        return p;
    };
    bf16* xb   = (bf16*)alloc((size_t)M * DIM_ * 2);
    bf16* wqb  = (bf16*)alloc((size_t)DIM_ * DIM_ * 2);
    bf16* wkb  = (bf16*)alloc((size_t)NKV_ * HD_ * DIM_ * 2);
    bf16* wvb  = (bf16*)alloc((size_t)NKV_ * HD_ * DIM_ * 2);
    bf16* wob  = (bf16*)alloc((size_t)DIM_ * DIM_ * 2);
    bf16* qraw = (bf16*)alloc((size_t)M * NH_ * HD_ * 2);
    bf16* kraw = (bf16*)alloc((size_t)M * NKV_ * HD_ * 2);
    bf16* vb   = (bf16*)alloc((size_t)M * NKV_ * HD_ * 2);
    bf16* qb   = (bf16*)alloc((size_t)M * NH_ * HD_ * 2);
    bf16* kb   = (bf16*)alloc((size_t)M * NKV_ * HD_ * 2);
    bf16* ob   = (bf16*)alloc((size_t)M * NH_ * HD_ * 2);

    cvt_f32_to_bf16<<<512, 256, 0, stream>>>(x,  xb,  (long long)M * DIM_);
    cvt_f32_to_bf16<<<512, 256, 0, stream>>>(wq, wqb, (long long)DIM_ * DIM_);
    cvt_f32_to_bf16<<<256, 256, 0, stream>>>(wk, wkb, (long long)NKV_ * HD_ * DIM_);
    cvt_f32_to_bf16<<<256, 256, 0, stream>>>(wv, wvb, (long long)NKV_ * HD_ * DIM_);
    cvt_f32_to_bf16<<<512, 256, 0, stream>>>(wo, wob, (long long)DIM_ * DIM_);

    // Projections (C = A * W^T)
    gemm_nt_bf16<true><<<dim3(DIM_ / 128, M / 128), 256, 0, stream>>>(
        xb, wqb, qraw, M, DIM_, DIM_);
    gemm_nt_bf16<true><<<dim3((NKV_ * HD_) / 128, M / 128), 256, 0, stream>>>(
        xb, wkb, kraw, M, NKV_ * HD_, DIM_);
    gemm_nt_bf16<true><<<dim3((NKV_ * HD_) / 128, M / 128), 256, 0, stream>>>(
        xb, wvb, vb, M, NKV_ * HD_, DIM_);

    // RMSNorm + RoPE on Q and K
    rmsnorm_rope_kernel<<<M * NH_, 64, 0, stream>>>(qraw, qnw, fr, qb, NH_);
    rmsnorm_rope_kernel<<<M * NKV_, 64, 0, stream>>>(kraw, knw, fr, kb, NKV_);

    // Causal GQA flash attention
    flash_attn_kernel<<<dim3(SEQ_ / 128, NH_, BS_), 256, 0, stream>>>(qb, kb, vb, ob);

    // Output projection (fp32 result)
    gemm_nt_bf16<false><<<dim3(DIM_ / 128, M / 128), 256, 0, stream>>>(
        ob, wob, out, M, DIM_, DIM_);
}
